// MultiHeadAttention_16028817949043
// MI455X (gfx1250) — matrix operand order repaired and checked by reading, not compiled, not run on hardware
//
#include <hip/hip_runtime.h>

#define BB 4
#define SS 1024
#define DD 1024
#define HH 16
#define DKC 64
#define OUT0 ((size_t)BB * SS * DD)   // 4,194,304 floats of output before weights

typedef __attribute__((ext_vector_type(16))) __bf16 v16bf;
typedef __attribute__((ext_vector_type(8)))  __bf16 v8bf;
typedef __attribute__((ext_vector_type(8)))  float  v8f;

__device__ __forceinline__ v8f wmma_bf16(v16bf a, v16bf b, v8f c) {
  // (neg_a, A, neg_b, B, c_mod, C, reuse_a, reuse_b)
  return __builtin_amdgcn_wmma_f32_16x16x32_bf16(false, a, false, b, (short)0, c,
                                                 false, false);
}

// A fragment (16x32 bf16): p = &src[(row0 + (lane&15))*ld + k0 + ((lane>>4)<<4)]
__device__ __forceinline__ v16bf load_a_frag(const __bf16* p) {
  v8bf lo = *(const v8bf*)(p);
  v8bf hi = *(const v8bf*)(p + 8);
  return __builtin_shufflevector(lo, hi, 0, 1, 2, 3, 4, 5, 6, 7,
                                         8, 9, 10, 11, 12, 13, 14, 15);
}

// B fragment (32x16 bf16) from N-major source bt[n][k]:
// p = &bt[(n0 + (lane&15))*ld + k0 + ((lane>>4)<<4)]; 16 contiguous K values.
__device__ __forceinline__ v16bf load_b_frag(const __bf16* p) {
  return *(const v16bf*)p;
}

// ---- gfx1250 async global->LDS copy (ASYNCcnt path) via inline asm ---------
// vdst = LDS byte address (low 32 bits of flat shared pointer), vaddr = 64-bit
// global address, saddr = off (NULL).
__device__ __forceinline__ void async_copy_b128(const __bf16* g, __bf16* l) {
  const unsigned lds = (unsigned)(unsigned long long)l;
  const unsigned long long ga = (unsigned long long)g;
  asm volatile("global_load_async_to_lds_b128 %0, %1, off"
               :: "v"(lds), "v"(ga) : "memory");
}

__device__ __forceinline__ void wait_async0() {
#if __has_builtin(__builtin_amdgcn_s_wait_asynccnt)
  __builtin_amdgcn_s_wait_asynccnt(0);
#else
  asm volatile("s_wait_asynccnt 0x0" ::: "memory");
#endif
}

// ---------------------------------------------------------------------------
// Kernel 1: fp32 -> bf16 packing / layout transforms
// ---------------------------------------------------------------------------
__global__ void __launch_bounds__(256) convert_kernel(
    const float* __restrict__ x,  const float* __restrict__ WQ,
    const float* __restrict__ WK, const float* __restrict__ WV,
    const float* __restrict__ WO, const float* __restrict__ bQ,
    const float* __restrict__ bK, const float* __restrict__ bV,
    __bf16* __restrict__ xb, __bf16* __restrict__ wt,
    float* __restrict__ bqkv, __bf16* __restrict__ wot) {
  const int i = blockIdx.x * 256 + threadIdx.x;          // [0, 4M)
  if (i < BB * SS * DD) xb[i] = (__bf16)x[i];
  if (i < 3 * DD * DD) {                                 // wt[c][d] = W[seg][h][d][k]
    const int c = i >> 10, d = i & 1023;
    const int seg = c >> 10, cc = c & 1023, h = cc >> 6, k = cc & 63;
    const float* W = (seg == 0) ? WQ : ((seg == 1) ? WK : WV);
    wt[i] = (__bf16)W[((size_t)h * DD + d) * DKC + k];
  }
  if (i < DD * DD) {                                     // wot[j][d] = WO[d][j]
    const int j = i >> 10, d = i & 1023;
    wot[i] = (__bf16)WO[(size_t)d * DD + j];
  }
  if (i < 3 * DD)
    bqkv[i] = (i < DD) ? bQ[i] : ((i < 2 * DD) ? bK[i - DD] : bV[i - 2 * DD]);
}

// ---------------------------------------------------------------------------
// Shared GEMM mainloop: 128(M) x 64(N) block tile, LDS-staged double-buffered
// B (64 x 64K bf16 per stage, async copies), register-pipelined A fragments.
// acc[4] = per-wave 16x64 tile.
// ---------------------------------------------------------------------------
__device__ __forceinline__ void gemm_mainloop_128x64(
    const __bf16* __restrict__ A, const __bf16* __restrict__ Bt,
    __bf16* bs /* LDS [2][64][64] */, v8f acc[4]) {
  const int tid  = threadIdx.x;
  const int lane = tid & 31;
  const int wave = tid >> 5;
  const int lh   = lane >> 4;
  const int nlo  = lane & 15;
  const int m0   = blockIdx.x * 128 + wave * 16;
  const int n0   = blockIdx.y * 64;

  // async-copy assignment: thread copies 32B of the stage: row=tid>>2,
  // col=(tid&3)*16 (bf16 elements); whole 8KB stage = 256 thr * 2 * b128.
  const int crow = tid >> 2;
  const int ccol = (tid & 3) << 4;
  const __bf16* gsrc = Bt + (size_t)(n0 + crow) * DD + ccol;
  __bf16* lbase = bs + crow * 64 + ccol;

  // stage 0 copies
  async_copy_b128(gsrc, lbase);
  async_copy_b128(gsrc + 8, lbase + 8);

  const __bf16* arow = A + (size_t)(m0 + nlo) * DD + (lh << 4);
  v16bf a0 = load_a_frag(arow);
  v16bf a1 = load_a_frag(arow + 32);

  for (int st = 0; st < 16; ++st) {   // 16 stages of K=64
    const int buf = st & 1;
    const __bf16* bsc = bs + buf * 4096;

    wait_async0();        // my stage-`st` copies complete
    __syncthreads();      // everyone's complete; prior stage's ds reads done

    if (st + 1 < 16) {    // overlap copy of stage st+1 with compute of st
      const __bf16* gn = gsrc + (st + 1) * 64;
      __bf16* ln = bs + (buf ^ 1) * 4096 + crow * 64 + ccol;
      async_copy_b128(gn, ln);
      async_copy_b128(gn + 8, ln + 8);
    }
    v16bf na0 = a0, na1 = a1;
    if (st + 1 < 16) {    // register-pipeline A
      na0 = load_a_frag(arow + (st + 1) * 64);
      na1 = load_a_frag(arow + (st + 1) * 64 + 32);
    }
#pragma unroll
    for (int j = 0; j < 4; ++j) {
      const __bf16* bp = bsc + (j * 16 + nlo) * 64 + (lh << 4);
      acc[j] = wmma_bf16(a0, load_b_frag(bp), acc[j]);
      acc[j] = wmma_bf16(a1, load_b_frag(bp + 32), acc[j]);
    }
    a0 = na0;
    a1 = na1;
  }
}

// ---------------------------------------------------------------------------
// Kernel 2: QKV projection GEMM  [4096 x 1024] @ [1024 x 3072]
// ---------------------------------------------------------------------------
__global__ void __launch_bounds__(256) gemm_qkv_kernel(
    const __bf16* __restrict__ xb, const __bf16* __restrict__ wt,
    const float* __restrict__ bqkv, __bf16* __restrict__ Qb,
    __bf16* __restrict__ Kb, __bf16* __restrict__ Vt) {
  __shared__ __align__(32) __bf16 bs[2][64][64];
  const int lane = threadIdx.x & 31;
  const int wave = threadIdx.x >> 5;
  const int lh   = lane >> 4;
  const int nlo  = lane & 15;
  const int m0   = blockIdx.x * 128 + wave * 16;
  const int n0   = blockIdx.y * 64;

  v8f acc[4] = {};
  gemm_mainloop_128x64(xb, wt, &bs[0][0][0], acc);

#pragma unroll
  for (int j = 0; j < 4; ++j) {
    const int c   = n0 + j * 16 + nlo;
    const int seg = c >> 10, cc = c & 1023, h = cc >> 6, kk = cc & 63;
#pragma unroll
    for (int r = 0; r < 8; ++r) {
      const int m = m0 + r + (lh << 3);
      const int b = m >> 10, s = m & 1023;
      const float v = acc[j][r] + bqkv[c];
      if (seg == 2) {
        Vt[((size_t)((b * HH + h) * DKC + kk)) * SS + s] = (__bf16)v;
      } else {
        const size_t off = ((size_t)((b * HH + h) * SS + s)) * DKC + kk;
        if (seg == 0) Qb[off] = (__bf16)v; else Kb[off] = (__bf16)v;
      }
    }
  }
}

// ---------------------------------------------------------------------------
// Kernel 3: attention per (b, h, 16-query tile)
// ---------------------------------------------------------------------------
__global__ void __launch_bounds__(256) attention_kernel(
    const __bf16* __restrict__ Qb, const __bf16* __restrict__ Kb,
    const __bf16* __restrict__ Vt, float* __restrict__ out,
    __bf16* __restrict__ heads) {
  __shared__ __align__(32) __bf16 P[16][SS];    // 32 KB; later reused as partials
  __shared__ float redmax[16][8];
  __shared__ float redsum[16][8];

  const int lane = threadIdx.x & 31;
  const int wave = threadIdx.x >> 5;
  const int lh   = lane >> 4;
  const int nlo  = lane & 15;
  const int qt   = blockIdx.x & 63;
  const int bh   = blockIdx.x >> 6;
  const int h    = bh & (HH - 1);
  const int b    = bh >> 4;
  const int q0   = qt * 16;

  const __bf16* Qbase = Qb + (size_t)bh * SS * DKC;
  const __bf16* Kbase = Kb + (size_t)bh * SS * DKC;
  const __bf16* Vbase = Vt + (size_t)bh * DKC * SS;

  // ---- Phase 1: scores = Q K^T / 8; wave owns columns [wave*128, +128)
  const __bf16* aq = Qbase + (size_t)(q0 + nlo) * DKC + (lh << 4);
  v16bf a0 = load_a_frag(aq);
  v16bf a1 = load_a_frag(aq + 32);

  const __bf16* kb0 = Kbase + (size_t)(wave * 128 + nlo) * DKC + (lh << 4);
  v16bf b0 = load_b_frag(kb0);
  v16bf b1 = load_b_frag(kb0 + 32);
  v8f sc[8];
#pragma unroll
  for (int j = 0; j < 8; ++j) {
    v16bf nb0 = b0, nb1 = b1;
    if (j < 7) {                         // register-pipeline next K tile
      const __bf16* kb = kb0 + (size_t)(j + 1) * 16 * DKC;
      nb0 = load_b_frag(kb);
      nb1 = load_b_frag(kb + 32);
    }
    v8f c = {};
    c = wmma_bf16(a0, b0, c);
    c = wmma_bf16(a1, b1, c);
    sc[j] = c * 0.125f;   // 1/sqrt(64)
    b0 = nb0;
    b1 = nb1;
  }

  // ---- Phase 2: softmax across 1024 columns (8 waves)
  float rmax[8], inv[8];
#pragma unroll
  for (int r = 0; r < 8; ++r) {
    float m = sc[0][r];
#pragma unroll
    for (int j = 1; j < 8; ++j) m = fmaxf(m, sc[j][r]);
#pragma unroll
    for (int msk = 1; msk < 16; msk <<= 1) m = fmaxf(m, __shfl_xor(m, msk, 32));
    rmax[r] = m;
  }
  if (nlo == 0) {
#pragma unroll
    for (int r = 0; r < 8; ++r) redmax[r + (lh << 3)][wave] = rmax[r];
  }
  __syncthreads();
#pragma unroll
  for (int r = 0; r < 8; ++r) {
    const int row = r + (lh << 3);
    float g = redmax[row][0];
#pragma unroll
    for (int w2 = 1; w2 < 8; ++w2) g = fmaxf(g, redmax[row][w2]);
    rmax[r] = g;
  }
#pragma unroll
  for (int r = 0; r < 8; ++r) {
    float s = 0.0f;
#pragma unroll
    for (int j = 0; j < 8; ++j) {
      const float e = __expf(sc[j][r] - rmax[r]);
      sc[j][r] = e;
      s += e;
    }
#pragma unroll
    for (int msk = 1; msk < 16; msk <<= 1) s += __shfl_xor(s, msk, 32);
    if (nlo == 0) redsum[r + (lh << 3)][wave] = s;
  }
  __syncthreads();
#pragma unroll
  for (int r = 0; r < 8; ++r) {
    const int row = r + (lh << 3);
    float s = 0.0f;
#pragma unroll
    for (int w2 = 0; w2 < 8; ++w2) s += redsum[row][w2];
    inv[r] = 1.0f / s;
  }

  // weights out (fp32, required output #2) + P tile (bf16, LDS) for PV
  float* wbase = out + OUT0 + (size_t)bh * SS * SS + (size_t)q0 * SS;
#pragma unroll
  for (int j = 0; j < 8; ++j) {
    const int col = wave * 128 + j * 16 + nlo;
#pragma unroll
    for (int r = 0; r < 8; ++r) {
      const int row = r + (lh << 3);
      const float wv = sc[j][r] * inv[r];
      wbase[(size_t)row * SS + col] = wv;
      P[row][col] = (__bf16)wv;
    }
  }
  __syncthreads();

  // ---- Phase 3: heads = P @ V ; wave owns K(s)-range [wave*128, +128)
  v8f pacc[4] = {};
  v16bf vb[4];
#pragma unroll
  for (int j = 0; j < 4; ++j)
    vb[j] = load_b_frag(Vbase + (size_t)(j * 16 + nlo) * SS + wave * 128 +
                        (lh << 4));
#pragma unroll
  for (int kk = 0; kk < 4; ++kk) {
    const int sbase = wave * 128 + kk * 32;
    v16bf nvb[4];
#pragma unroll
    for (int j = 0; j < 4; ++j) nvb[j] = vb[j];
    if (kk < 3) {                       // register-pipeline next V tiles
#pragma unroll
      for (int j = 0; j < 4; ++j)
        nvb[j] = load_b_frag(Vbase + (size_t)(j * 16 + nlo) * SS + sbase + 32 +
                             (lh << 4));
    }
    v16bf a = load_a_frag(&P[nlo][sbase + (lh << 4)]);   // ds_load_b128 x2
#pragma unroll
    for (int j = 0; j < 4; ++j) pacc[j] = wmma_bf16(a, vb[j], pacc[j]);
#pragma unroll
    for (int j = 0; j < 4; ++j) vb[j] = nvb[j];
  }
  __syncthreads();                 // all reads of P done -> reuse as partials
  float* part = (float*)P;         // [8 waves][16 rows][64 cols] = 32 KB
#pragma unroll
  for (int j = 0; j < 4; ++j)
#pragma unroll
    for (int r = 0; r < 8; ++r)
      part[((wave * 16) + r + (lh << 3)) * DKC + j * 16 + nlo] = pacc[j][r];
  __syncthreads();

  for (int i = threadIdx.x; i < 16 * DKC; i += 256) {
    const int r = i >> 6, cc = i & 63;
    float v = 0.0f;
#pragma unroll
    for (int w2 = 0; w2 < 8; ++w2) v += part[(w2 * 16 + r) * DKC + cc];
    heads[((size_t)(b * SS + q0 + r)) * DD + h * DKC + cc] = (__bf16)v;
  }
}

// ---------------------------------------------------------------------------
// Kernel 4: output projection GEMM  [4096 x 1024] @ [1024 x 1024] + bO
// ---------------------------------------------------------------------------
__global__ void __launch_bounds__(256) gemm_out_kernel(
    const __bf16* __restrict__ heads, const __bf16* __restrict__ wot,
    const float* __restrict__ bO, float* __restrict__ out) {
  __shared__ __align__(32) __bf16 bs[2][64][64];
  const int lane = threadIdx.x & 31;
  const int wave = threadIdx.x >> 5;
  const int lh   = lane >> 4;
  const int nlo  = lane & 15;
  const int m0   = blockIdx.x * 128 + wave * 16;
  const int n0   = blockIdx.y * 64;

  v8f acc[4] = {};
  gemm_mainloop_128x64(heads, wot, &bs[0][0][0], acc);

#pragma unroll
  for (int j = 0; j < 4; ++j) {
    const int c = n0 + j * 16 + nlo;
#pragma unroll
    for (int r = 0; r < 8; ++r) {
      const int m = m0 + r + (lh << 3);
      out[(size_t)m * DD + c] = acc[j][r] + bO[c];
    }
  }
}

// ---------------------------------------------------------------------------
extern "C" void kernel_launch(void* const* d_in, const int* in_sizes, int n_in,
                              void* d_out, int out_size, void* d_ws,
                              size_t ws_size, hipStream_t stream) {
  const float* x  = (const float*)d_in[0];
  const float* WQ = (const float*)d_in[1];
  const float* bQ = (const float*)d_in[2];
  const float* WK = (const float*)d_in[3];
  const float* bK = (const float*)d_in[4];
  const float* WV = (const float*)d_in[5];
  const float* bV = (const float*)d_in[6];
  const float* WO = (const float*)d_in[7];
  const float* bO = (const float*)d_in[8];
  float* out = (float*)d_out;

  size_t off = 0;
  auto carve = [&](size_t bytes) -> char* {
    char* p = (char*)d_ws + off;
    off += (bytes + 255) & ~(size_t)255;
    return p;
  };
  __bf16* xb    = (__bf16*)carve((size_t)BB * SS * DD * 2);        // 8 MB
  __bf16* wt    = (__bf16*)carve((size_t)3 * DD * DD * 2);         // 6 MB
  float*  bqkv  = (float*) carve((size_t)3 * DD * 4);
  __bf16* wot   = (__bf16*)carve((size_t)DD * DD * 2);             // 2 MB
  __bf16* Qb    = (__bf16*)carve((size_t)BB * HH * SS * DKC * 2);  // 8 MB
  __bf16* Kb    = (__bf16*)carve((size_t)BB * HH * SS * DKC * 2);  // 8 MB
  __bf16* Vt    = (__bf16*)carve((size_t)BB * HH * SS * DKC * 2);  // 8 MB
  __bf16* heads = (__bf16*)carve((size_t)BB * SS * DD * 2);        // 8 MB

  convert_kernel<<<16384, 256, 0, stream>>>(x, WQ, WK, WV, WO, bQ, bK, bV,
                                            xb, wt, bqkv, wot);
  gemm_qkv_kernel<<<dim3(32, 48), 256, 0, stream>>>(xb, wt, bqkv, Qb, Kb, Vt);
  attention_kernel<<<BB * HH * (SS / 16), 256, 0, stream>>>(Qb, Kb, Vt, out,
                                                            heads);
  gemm_out_kernel<<<dim3(32, 16), 256, 0, stream>>>(heads, wot, bO, out);
}
